// DavisManifold_25615184953464
// MI455X (gfx1250) — matrix operand order repaired and checked by reading, not compiled, not run on hardware
//
#include <hip/hip_runtime.h>

#define DIM 64
#define HID 128
#define NCOMP 2080
#define NTILES (NCOMP / 16)        // 130, exact
#define TCHUNK 2
#define WAVES 8
#define PTS_PER_BLOCK 128          // 8 waves x 16 points
#define BLOCK 256
#define TILE_B (16 * HID * 2)      // one W2 tile: 16 cols x 128 K f16 = 4096 bytes

typedef _Float16 v16h __attribute__((ext_vector_type(16)));
typedef _Float16 v8h  __attribute__((ext_vector_type(8)));
typedef float    v8f  __attribute__((ext_vector_type(8)));

// ---------------- shared memory layout (bytes, all 16B aligned) ----------------
#define SM_W1T   0                                   // HID*DIM f16      = 16384
#define SM_B1    (SM_W1T + HID*DIM*2)                // 128 f32          = 512
#define SM_B2    (SM_B1 + HID*4)                     // 2080 f32         = 8320
#define SM_ROWS  (SM_B2 + NCOMP*4)                   // 2080 u8
#define SM_COLS  (SM_ROWS + NCOMP)                   // 2080 u8
#define SM_XS    (SM_COLS + NCOMP)                   // 128*64 f32       = 32768
#define SM_VS    (SM_XS + PTS_PER_BLOCK*DIM*4)       // 128*64 f32       = 32768
#define SM_H     (SM_VS + PTS_PER_BLOCK*DIM*4)       // 8*2*16*128 f16   = 65536
#define SM_W     (SM_H + WAVES*TCHUNK*16*HID*2)      // 8*2*16*64 f32    = 65536
#define SM_W2T   (SM_W + WAVES*TCHUNK*16*DIM*4)      // 2 x 4096 (double buffer)
#define SM_TOTAL (SM_W2T + 2*TILE_B)                 // 234176 bytes

__device__ __forceinline__ v16h pack16(v8h lo, v8h hi) {
  v16h a;
#pragma unroll
  for (int e = 0; e < 8; ++e) { a[e] = lo[e]; a[e + 8] = hi[e]; }
  return a;
}

// A fragment (16x32 f16), source row-major [m][k], stride in f16 elements.
// ISA layout: lanes 0-15 m=0..15 / lanes 16-31 m=0..15; elem e<8 -> K=kbase+8*half+e,
// elem e>=8 -> K=kbase+16+8*half+(e-8).  Two contiguous 16B chunks per lane.
__device__ __forceinline__ v16h ld_afrag(const _Float16* base, int stride, int m, int hf, int kbase) {
  const _Float16* row = base + m * stride + kbase + 8 * hf;
  v8h lo = *(const v8h*)(row);
  v8h hi = *(const v8h*)(row + 16);
  return pack16(lo, hi);
}

// B fragment (32x16 f16) from an [n][k] (k-contiguous) tile.
// elem e<8 -> K = kbase + 8*half + e, elem e>=8 -> K = kbase + 16 + 8*half + (e-8)  (two 16B chunks).
__device__ __forceinline__ v16h ld_bfrag(const _Float16* base, int stride, int n, int hf, int kbase) {
  const _Float16* row = base + n * stride + kbase + 8 * hf;
  v8h lo = *(const v8h*)(row);
  v8h hi = *(const v8h*)(row + 16);
  return pack16(lo, hi);
}

__device__ __forceinline__ float softplus_f(float xv) {
  return (xv > 20.0f) ? xv : log1pf(expf(xv));
}

// CDNA5 async global -> LDS copy, 16 bytes per enabled lane, tracked by ASYNCcnt.
// GVS mode: mem_addr = SGPR_base + VGPR_i32_offset; LDS dest = per-lane byte offset
// within the workgroup LDS allocation (dynamic LDS starts at offset 0 here).
__device__ __forceinline__ void async_copy_b128(unsigned int lds_off,
                                                unsigned long long gbase,
                                                unsigned int goff) {
  asm volatile("global_load_async_to_lds_b128 %0, %1, %2"
               :: "v"(lds_off), "v"(goff), "s"(gbase)
               : "memory");
}
__device__ __forceinline__ void wait_asynccnt0() {
  asm volatile("s_wait_asynccnt 0x0" ::: "memory");
}

// Transpose + f16-convert W2: [128][2080] f32  ->  [2080][128] f16 (k-contiguous)
__global__ __launch_bounds__(BLOCK) void davis_prep_w2(const float* __restrict__ W2,
                                                       _Float16* __restrict__ W2t) {
  int idx = blockIdx.x * BLOCK + threadIdx.x;
  if (idx < HID * NCOMP) {
    int k = idx / NCOMP;
    int c = idx % NCOMP;
    W2t[c * HID + k] = (_Float16)W2[idx];
  }
}

__global__ __launch_bounds__(BLOCK) void davis_distance(
    const float* __restrict__ x, const float* __restrict__ y,
    const float* __restrict__ W1, const float* __restrict__ b1,
    const float* __restrict__ b2, const _Float16* __restrict__ W2t,
    const int* __restrict__ nsteps_p, float* __restrict__ out, int Btot) {
  extern __shared__ __align__(16) char smem[];
  _Float16*      sW1t = (_Float16*)(smem + SM_W1T);
  float*         sB1  = (float*)(smem + SM_B1);
  float*         sB2  = (float*)(smem + SM_B2);
  unsigned char* sRow = (unsigned char*)(smem + SM_ROWS);
  unsigned char* sCol = (unsigned char*)(smem + SM_COLS);
  float*         sXS  = (float*)(smem + SM_XS);
  float*         sVS  = (float*)(smem + SM_VS);
  _Float16*      sH   = (_Float16*)(smem + SM_H);
  float*         sW   = (float*)(smem + SM_W);

  const int tid  = threadIdx.x;
  const int wave = tid >> 5;
  const int lane = tid & 31;
  const int m    = lane & 15;   // point row within wave tile / A row
  const int hf   = lane >> 4;   // lane half, per WMMA layouts
  const int ns   = *nsteps_p;
  const float inv_ns = 1.0f / (float)ns;
  const unsigned long long w2base = (unsigned long long)(uintptr_t)W2t;

  // ---- stage constants: W1^T (f16), biases, lower-tri index tables ----
  for (int idx = tid; idx < HID * DIM; idx += BLOCK) {
    int n = idx / DIM, k = idx % DIM;
    sW1t[idx] = (_Float16)W1[k * HID + n];
  }
  for (int idx = tid; idx < HID; idx += BLOCK) sB1[idx] = b1[idx];
  for (int idx = tid; idx < NCOMP; idx += BLOCK) sB2[idx] = b2[idx];
  for (int c = tid; c < NCOMP; c += BLOCK) {
    int i = (int)((sqrtf(8.0f * (float)c + 1.0f) - 1.0f) * 0.5f);
    while ((i + 1) * (i + 2) / 2 <= c) ++i;
    while (i * (i + 1) / 2 > c) --i;
    sRow[c] = (unsigned char)i;
    sCol[c] = (unsigned char)(c - i * (i + 1) / 2);
  }
  // ---- stage this block's x and v = (y-x)/n, grouped per wave ----
  for (int idx = tid; idx < PTS_PER_BLOCK * DIM; idx += BLOCK) {
    int pl = idx / DIM, k = idx % DIM;
    int g  = blockIdx.x * PTS_PER_BLOCK + pl;
    int gc = (g < Btot) ? g : (Btot - 1);
    float xv = x[gc * DIM + k];
    float yv = y[gc * DIM + k];
    int wv = pl >> 4, ml = pl & 15;
    sXS[(wv * 16 + ml) * DIM + k] = xv;
    sVS[(wv * 16 + ml) * DIM + k] = (yv - xv) * inv_ns;
  }
  __syncthreads();

  const float* xsw = sXS + wave * 16 * DIM;
  const float* vsw = sVS + wave * 16 * DIM;
  _Float16*    hw  = sH + wave * TCHUNK * 16 * HID;
  float*       ww  = sW + wave * TCHUNK * 16 * DIM;

  float vv = 0.0f;
  if (lane < 16) {
    for (int k = 0; k < DIM; ++k) { float v = vsw[lane * DIM + k]; vv += v * v; }
  }

  float total = 0.0f;

  for (int t0 = 0; t0 < ns; t0 += TCHUNK) {
    const int tc = (ns - t0 < TCHUNK) ? (ns - t0) : TCHUNK;  // wave-uniform

    // zero this wave's w = L^T v accumulators
    for (int idx = lane; idx < TCHUNK * 16 * DIM; idx += 32) ww[idx] = 0.0f;

    // ---- Phase A: h_t = relu(p_t @ W1 + b1) -> f16 in LDS, per step in chunk ----
#pragma unroll
    for (int tl = 0; tl < TCHUNK; ++tl) {
      if (tl < tc) {
        const float tf = (float)(t0 + tl);
        v8h lo, hi;
#pragma unroll
        for (int e = 0; e < 8; ++e) {
          int k0 = 8 * hf + e, k1 = 16 + 8 * hf + e;
          lo[e] = (_Float16)(xsw[m * DIM + k0] + tf * vsw[m * DIM + k0]);
          hi[e] = (_Float16)(xsw[m * DIM + k1] + tf * vsw[m * DIM + k1]);
        }
        v16h a0 = pack16(lo, hi);
#pragma unroll
        for (int e = 0; e < 8; ++e) {
          int k0 = 32 + 8 * hf + e, k1 = 48 + 8 * hf + e;
          lo[e] = (_Float16)(xsw[m * DIM + k0] + tf * vsw[m * DIM + k0]);
          hi[e] = (_Float16)(xsw[m * DIM + k1] + tf * vsw[m * DIM + k1]);
        }
        v16h a1 = pack16(lo, hi);
#pragma unroll
        for (int nt = 0; nt < HID / 16; ++nt) {
          const int n = nt * 16 + (lane & 15);
          v16h bf0 = ld_bfrag(sW1t, DIM, n, hf, 0);
          v16h bf1 = ld_bfrag(sW1t, DIM, n, hf, 32);
          v8f acc = {};
          acc = __builtin_amdgcn_wmma_f32_16x16x32_f16(false, a0, false, bf0, (short)0, acc, false, false);
          acc = __builtin_amdgcn_wmma_f32_16x16x32_f16(false, a1, false, bf1, (short)0, acc, false, false);
          const float bias = sB1[n];
#pragma unroll
          for (int r = 0; r < 8; ++r) {
            const int mm = r + 8 * hf;
            float hv = acc[r] + bias;
            hv = hv > 0.0f ? hv : 0.0f;
            hw[(tl * 16 + mm) * HID + n] = (_Float16)hv;
          }
        }
      }
    }

    // preload h A-fragments (reused across all 130 W2 tiles)
    v16h af[TCHUNK][4];
#pragma unroll
    for (int tl = 0; tl < TCHUNK; ++tl)
#pragma unroll
      for (int ks = 0; ks < 4; ++ks)
        af[tl][ks] = ld_afrag(hw + tl * 16 * HID, HID, m, hf, ks * 32);

    // ---- Phase B: double-buffered async staging of W2 tiles + GEMM2 + fold ----
    // Prologue: async-prefetch tile 0 into buffer 0 (16B per thread = 4KB tile).
    async_copy_b128((unsigned int)(SM_W2T + tid * 16), w2base,
                    (unsigned int)(tid * 16));
    wait_asynccnt0();
    __syncthreads();

    for (int nt = 0; nt < NTILES; ++nt) {
      const int cur = nt & 1;
      // prefetch next tile into the alternate buffer while computing this one
      if (nt + 1 < NTILES) {
        async_copy_b128((unsigned int)(SM_W2T + (cur ^ 1) * TILE_B + tid * 16), w2base,
                        (unsigned int)((nt + 1) * TILE_B + tid * 16));
      }
      const _Float16* sW2c = (const _Float16*)(smem + SM_W2T + cur * TILE_B);

      const int nl  = lane & 15;
      const int c   = nt * 16 + nl;        // global comp column
      const int ic  = sRow[c];             // L row index i
      const int jc  = sCol[c];             // L col index j
      const float b2v = sB2[c];
      const bool diag = (ic == jc);

      v16h bf[4];
#pragma unroll
      for (int ks = 0; ks < 4; ++ks) bf[ks] = ld_bfrag(sW2c, HID, nl, hf, ks * 32);

#pragma unroll
      for (int tl = 0; tl < TCHUNK; ++tl) {
        if (tl < tc) {
          v8f acc = {};
#pragma unroll
          for (int ks = 0; ks < 4; ++ks)
            acc = __builtin_amdgcn_wmma_f32_16x16x32_f16(false, af[tl][ks], false, bf[ks],
                                                         (short)0, acc, false, false);
          float* wt = ww + tl * 16 * DIM;
#pragma unroll
          for (int r = 0; r < 8; ++r) {
            const int mm = r + 8 * hf;
            float val = acc[r] + b2v;                             // comp[m][c]
            float coeff = diag ? (softplus_f(val) + 0.1f) : val;  // L[i][j]
            atomicAdd(&wt[mm * DIM + jc], coeff * vsw[mm * DIM + ic]);  // w_j += L[i][j]*v_i
          }
        }
      }
      // prefetch must have landed (per-wave ASYNCcnt) before anyone reads it next iter
      wait_asynccnt0();
      __syncthreads();
    }

    // ---- Phase C: inner = ||w||^2 + 0.01*||v||^2 ; total += sqrt(max(inner, 1e-8)) ----
    if (lane < 16) {
#pragma unroll
      for (int tl = 0; tl < TCHUNK; ++tl) {
        if (tl < tc) {
          const float* wt = ww + (tl * 16 + lane) * DIM;
          float s = 0.0f;
          for (int j = 0; j < DIM; ++j) { float wj = wt[j]; s += wj * wj; }
          float inner = fmaxf(s + 0.01f * vv, 1e-8f);
          total += sqrtf(inner);
        }
      }
    }
    __syncthreads();
  }

  if (lane < 16) {
    int g = blockIdx.x * PTS_PER_BLOCK + wave * 16 + lane;
    if (g < Btot) out[g] = total;
  }
}

extern "C" void kernel_launch(void* const* d_in, const int* in_sizes, int n_in,
                              void* d_out, int out_size, void* d_ws, size_t ws_size,
                              hipStream_t stream) {
  const float* x  = (const float*)d_in[0];
  const float* y  = (const float*)d_in[1];
  const float* W1 = (const float*)d_in[2];
  const float* b1 = (const float*)d_in[3];
  const float* W2 = (const float*)d_in[4];
  const float* b2 = (const float*)d_in[5];
  const int*   ns = (const int*)d_in[6];
  float* out = (float*)d_out;
  const int Btot = in_sizes[0] / DIM;

  _Float16* W2t = (_Float16*)d_ws;  // 2080*128 f16 = 532,480 bytes

  const int tblocks = (HID * NCOMP + BLOCK - 1) / BLOCK;
  davis_prep_w2<<<tblocks, BLOCK, 0, stream>>>(W2, W2t);

  const int nblocks = (Btot + PTS_PER_BLOCK - 1) / PTS_PER_BLOCK;
  davis_distance<<<nblocks, BLOCK, SM_TOTAL, stream>>>(x, y, W1, b1, b2, W2t, ns, out, Btot);
}
